// NeuralODEBaseline_20598663152189
// MI455X (gfx1250) — matrix operand order repaired and checked by reading, not compiled, not run on hardware
//
#include <hip/hip_runtime.h>
#include <hip/hip_bf16.h>

#define B_ROWS 32768
#define IN_DIM 512
#define LAT    256
#define HID    1024
#define MT     32           // rows of the ODE state per workgroup
#define NSTEPS 20
#define HSTEP  0.005f       // (DT*N_STEPS)/NS_INT = 0.1/20

typedef __attribute__((ext_vector_type(16))) __bf16 v16bf;
typedef __attribute__((ext_vector_type(8)))  __bf16 v8bf;
typedef __attribute__((ext_vector_type(8)))  float  v8f;

union V16 { v16bf v; v8bf h[2]; };

static __device__ __forceinline__ __bf16 f2bf(float f) {
    return (__bf16)f;                      // native v_cvt path
}
static __device__ __forceinline__ float bfu2f(unsigned short s) {
    unsigned u = ((unsigned)s) << 16;
    return __builtin_bit_cast(float, u);
}
static __device__ __forceinline__ unsigned pk2(float a, float b) {
    unsigned short ua = __builtin_bit_cast(unsigned short, (__bf16)a);
    unsigned short ub = __builtin_bit_cast(unsigned short, (__bf16)b);
    return (unsigned)ua | ((unsigned)ub << 16);
}
static __device__ __forceinline__ v8f zero8() {
    v8f z = {0.f, 0.f, 0.f, 0.f, 0.f, 0.f, 0.f, 0.f};
    return z;
}
// One instruction prefetches one cacheline per EXEC lane: with lane*128B
// spread, a single global_prefetch_b8 (no scope modifier = WGP scope) pulls
// a 4KB span into the WGP$.
static __device__ __forceinline__ void wgp_prefetch(const void* p) {
    asm volatile("global_prefetch_b8 %0, off" :: "v"(p));
}

// Load/store a thread-owned row of 32 bf16 elems (64B, aligned) <-> f32 regs.
static __device__ __forceinline__ void load_row_bf16(const __bf16* src, float v[32]) {
    const uint4* p = (const uint4*)src;
    #pragma unroll
    for (int b = 0; b < 4; ++b) {
        uint4 q = p[b];
        unsigned w0 = q.x, w1 = q.y, w2 = q.z, w3 = q.w;
        v[b*8+0] = bfu2f((unsigned short)(w0 & 0xFFFFu));
        v[b*8+1] = bfu2f((unsigned short)(w0 >> 16));
        v[b*8+2] = bfu2f((unsigned short)(w1 & 0xFFFFu));
        v[b*8+3] = bfu2f((unsigned short)(w1 >> 16));
        v[b*8+4] = bfu2f((unsigned short)(w2 & 0xFFFFu));
        v[b*8+5] = bfu2f((unsigned short)(w2 >> 16));
        v[b*8+6] = bfu2f((unsigned short)(w3 & 0xFFFFu));
        v[b*8+7] = bfu2f((unsigned short)(w3 >> 16));
    }
}
static __device__ __forceinline__ void store_row_bf16(__bf16* dst, const float v[32]) {
    uint4* p = (uint4*)dst;
    #pragma unroll
    for (int b = 0; b < 4; ++b) {
        uint4 q;
        q.x = pk2(v[b*8+0], v[b*8+1]);
        q.y = pk2(v[b*8+2], v[b*8+3]);
        q.z = pk2(v[b*8+4], v[b*8+5]);
        q.w = pk2(v[b*8+6], v[b*8+7]);
        p[b] = q;
    }
}

// zarg = z + sum_s cf[s]*k[s]  (cf pre-multiplied by h), one k slot at a time.
template<int NS>
static __device__ __forceinline__ void make_zarg(
    __bf16* zargp, const __bf16* kbase, const float z[32], const float (&cf)[NS])
{
    float zv[32];
    #pragma unroll
    for (int j = 0; j < 32; ++j) zv[j] = z[j];
    #pragma unroll
    for (int s = 0; s < NS; ++s) {
        float kv[32];
        load_row_bf16(kbase + s * 8192, kv);
        #pragma unroll
        for (int j = 0; j < 32; ++j) zv[j] += cf[s] * kv[j];
    }
    store_row_bf16(zargp, zv);
}
template<int NS>
static __device__ __forceinline__ void update_z(
    float z[32], const __bf16* kbase, const float (&cf)[NS])
{
    #pragma unroll
    for (int s = 0; s < NS; ++s) {
        float kv[32];
        load_row_bf16(kbase + s * 8192, kv);
        #pragma unroll
        for (int j = 0; j < 32; ++j) z[j] += cf[s] * kv[j];
    }
}

// ---------------------------------------------------------------------------
// Pack an f32 KxN row-major weight into bf16 WMMA B-fragment order.
// 16-bit B 32x16: lane l holds column N = l%16, elem e at K = (e&7) + 8*(l/16) + 16*(e/8).
// out[((nt*Kt + kt)*32 + l)*16 + e] = W[kt*32 + (e&7) + 8*(l>>4) + 16*(e>>3)][nt*16 + (l&15)]
// ---------------------------------------------------------------------------
__global__ void pack_b_kernel(const float* __restrict__ W, __bf16* __restrict__ P,
                              int K, int N) {
    int idx = blockIdx.x * blockDim.x + threadIdx.x;
    if (idx >= K * N) return;
    int Kt   = K >> 5;
    int e    = idx & 15;
    int l    = (idx >> 4) & 31;
    int frag = idx >> 9;
    int kt   = frag % Kt;
    int nt   = frag / Kt;
    P[idx] = f2bf(W[(kt * 32 + (e & 7) + ((l >> 4) << 3) + ((e >> 3) << 4)) * N + nt * 16 + (l & 15)]);
}

// ---------------------------------------------------------------------------
// One wave computes G 16x16 C tiles for BOTH 16-row M blocks (B fragment
// reused across two A fragments -> halves global weight traffic).
// A stream uses one loop-carried LDS pointer: the 4 fragment loads per kt are
// immediate-offset ds_load_b128 (second M block at constant 16*ASTR elems,
// 8KB/32KB -> fits the 16-bit DS offset), so no per-kt address VALU.
// ---------------------------------------------------------------------------
template<int G, int KT, int ASTR>
static __device__ __forceinline__ void gemm_acc2(
    const __bf16* __restrict__ A, const __bf16* __restrict__ Bp,
    int ntile0, int lane, v8f acc0[G], v8f acc1[G])
{
    const int row  = lane & 15;
    const int ksel = (lane >> 4) << 3;
    const __bf16* ap   = A + row * ASTR + ksel;           // M block 0 (rows 0-15)
    const __bf16* breg = Bp + (size_t)ntile0 * KT * 512;
    const char*   bpf  = (const char*)breg + lane * 128;
    wgp_prefetch(bpf);
    wgp_prefetch(bpf + 4096);
    const __bf16* bbase = breg + lane * 16;
    #pragma unroll 4
    for (int kt = 0; kt < KT; ++kt) {
        V16 a0, a1;
        a0.h[0] = *(const v8bf*)(ap);
        a0.h[1] = *(const v8bf*)(ap + 16);
        a1.h[0] = *(const v8bf*)(ap + 16 * ASTR);         // M block 1 (rows 16-31)
        a1.h[1] = *(const v8bf*)(ap + 16 * ASTR + 16);
        ap += 32;
        #pragma unroll
        for (int g = 0; g < G; ++g) {
            const __bf16* bp = bbase + ((size_t)g * KT + kt) * 512;
            V16 b;
            b.h[0] = *(const v8bf*)(bp);
            b.h[1] = *(const v8bf*)(bp + 8);
            acc0[g] = __builtin_amdgcn_wmma_f32_16x16x32_bf16(
                false, a0.v, false, b.v, (short)0, acc0[g], false, false);
            acc1[g] = __builtin_amdgcn_wmma_f32_16x16x32_bf16(
                false, a1.v, false, b.v, (short)0, acc1[g], false, false);
        }
    }
}

// C layout: VGPR r, lanes 0-15 -> M=r, N=lane; lanes 16-31 -> M=r+8.
template<int G, int OSTR>
static __device__ __forceinline__ void store_silu_bf16(
    __bf16* __restrict__ O, const float* __restrict__ bias,
    int mblk16, int ntile0, int lane, const v8f acc[G])
{
    const int col16 = lane & 15;
    const int mbase = mblk16 + ((lane >> 4) << 3);
    #pragma unroll
    for (int g = 0; g < G; ++g) {
        const int col = (ntile0 + g) * 16 + col16;
        const float bb = bias[col];
        #pragma unroll
        for (int r = 0; r < 8; ++r) {
            float v = acc[g][r] + bb;
            v = v / (1.0f + __expf(-v));     // SiLU
            O[(mbase + r) * OSTR + col] = f2bf(v);
        }
    }
}
template<int G, int OSTR>
static __device__ __forceinline__ void store_bias_bf16(
    __bf16* __restrict__ O, const float* __restrict__ bias,
    int mblk16, int ntile0, int lane, const v8f acc[G])
{
    const int col16 = lane & 15;
    const int mbase = mblk16 + ((lane >> 4) << 3);
    #pragma unroll
    for (int g = 0; g < G; ++g) {
        const int col = (ntile0 + g) * 16 + col16;
        const float bb = bias[col];
        #pragma unroll
        for (int r = 0; r < 8; ++r)
            O[(mbase + r) * OSTR + col] = f2bf(acc[g][r] + bb);
    }
}
template<int G, int OSTR>
static __device__ __forceinline__ void store_f32(
    float* __restrict__ O, const float* __restrict__ bias,
    int mblk16, int ntile0, int lane, const v8f acc[G])
{
    const int col16 = lane & 15;
    const int mbase = mblk16 + ((lane >> 4) << 3);
    #pragma unroll
    for (int g = 0; g < G; ++g) {
        const int col = (ntile0 + g) * 16 + col16;
        const float bb = bias[col];
        #pragma unroll
        for (int r = 0; r < 8; ++r)
            O[(mbase + r) * OSTR + col] = acc[g][r] + bb;
    }
}

// ---------------------------------------------------------------------------
// One evaluation of f(z):  bufA holds Zarg (32x256 bf16).
// GEMM1 -> SiLU -> H1 in bufB (32x1024 bf16)
// GEMM2 -> SiLU -> H2 in bufA (32x1024 bf16)
// GEMM3 -> +b3  -> k slot (32x256 bf16)
// ---------------------------------------------------------------------------
static __device__ __forceinline__ void feval(
    __bf16* bufA, __bf16* bufB, __bf16* kslot,
    const __bf16* __restrict__ w1p, const float* __restrict__ b1,
    const __bf16* __restrict__ w2p, const float* __restrict__ b2,
    const __bf16* __restrict__ w3p, const float* __restrict__ b3,
    int wave, int lane)
{
    __syncthreads();   // Zarg visible, previous k-slot readers done
    #pragma unroll 1
    for (int grp = 0; grp < 2; ++grp) {
        v8f a0[4], a1[4];
        #pragma unroll
        for (int g = 0; g < 4; ++g) { a0[g] = zero8(); a1[g] = zero8(); }
        int nt0 = wave * 8 + grp * 4;
        gemm_acc2<4, 8, LAT>(bufA, w1p, nt0, lane, a0, a1);
        store_silu_bf16<4, HID>(bufB, b1, 0,  nt0, lane, a0);
        store_silu_bf16<4, HID>(bufB, b1, 16, nt0, lane, a1);
    }
    __syncthreads();
    #pragma unroll 1
    for (int grp = 0; grp < 2; ++grp) {
        v8f a0[4], a1[4];
        #pragma unroll
        for (int g = 0; g < 4; ++g) { a0[g] = zero8(); a1[g] = zero8(); }
        int nt0 = wave * 8 + grp * 4;
        gemm_acc2<4, 32, HID>(bufB, w2p, nt0, lane, a0, a1);
        store_silu_bf16<4, HID>(bufA, b2, 0,  nt0, lane, a0);
        store_silu_bf16<4, HID>(bufA, b2, 16, nt0, lane, a1);
    }
    __syncthreads();
    {
        v8f a0[2], a1[2];
        a0[0]=zero8(); a0[1]=zero8(); a1[0]=zero8(); a1[1]=zero8();
        int nt0 = wave * 2;
        gemm_acc2<2, 32, HID>(bufA, w3p, nt0, lane, a0, a1);
        store_bias_bf16<2, LAT>(kslot, b3, 0,  nt0, lane, a0);
        store_bias_bf16<2, LAT>(kslot, b3, 16, nt0, lane, a1);
    }
    __syncthreads();
}

// ---------------------------------------------------------------------------
// Fused persistent-tile Neural ODE: 1024 blocks x 256 threads (8 waves).
// LDS: bufA 64KB + bufB 64KB + 6 bf16 k-slots 96KB = 224KB (CDNA5 320KB WGP).
// Per-thread persistent state: 32 f32 elems of z (row tid>>3, cols (tid&7)*32..).
// ---------------------------------------------------------------------------
__global__ __launch_bounds__(256) void node_fused_kernel(
    const float* __restrict__ x,
    const __bf16* __restrict__ encp, const float* __restrict__ enc_b,
    const __bf16* __restrict__ w1p,  const float* __restrict__ b1,
    const __bf16* __restrict__ w2p,  const float* __restrict__ b2,
    const __bf16* __restrict__ w3p,  const float* __restrict__ b3,
    const __bf16* __restrict__ decp, const float* __restrict__ dec_b,
    float* __restrict__ out)
{
    __shared__ __align__(16) unsigned char sA[65536];
    __shared__ __align__(16) unsigned char sB[65536];
    __shared__ __align__(16) unsigned char sK[98304];
    __bf16* bufA = (__bf16*)sA;
    __bf16* bufB = (__bf16*)sB;
    __bf16* kb   = (__bf16*)sK;          // 6 slots of 8192 bf16 (32x256)

    const int tid  = threadIdx.x;
    const int wave = tid >> 5;
    const int lane = tid & 31;
    const int row0 = blockIdx.x * MT;

    // ---- stage x tile (32x512 f32) into bufA as packed bf16 ----
    {
        const float* xt = x + (size_t)row0 * IN_DIM;
        #pragma unroll
        for (int b = 0; b < 8; ++b) {
            int idx = tid * 64 + b * 8;
            float4 p0 = *(const float4*)(xt + idx);
            float4 p1 = *(const float4*)(xt + idx + 4);
            uint4 q;
            q.x = pk2(p0.x, p0.y); q.y = pk2(p0.z, p0.w);
            q.z = pk2(p1.x, p1.y); q.w = pk2(p1.z, p1.w);
            *(uint4*)(bufA + idx) = q;
        }
    }
    __syncthreads();

    // ---- encoder: (32x512) @ (512x256) + enc_b -> f32 scratch in sK ----
    {
        v8f a0[2], a1[2];
        a0[0]=zero8(); a0[1]=zero8(); a1[0]=zero8(); a1[1]=zero8();
        int nt0 = wave * 2;
        gemm_acc2<2, 16, IN_DIM>(bufA, encp, nt0, lane, a0, a1);
        store_f32<2, LAT>((float*)sK, enc_b, 0,  nt0, lane, a0);
        store_f32<2, LAT>((float*)sK, enc_b, 16, nt0, lane, a1);
    }
    __syncthreads();

    const int own = (tid >> 3) * LAT + (tid & 7) * 32;   // 32 owned elems
    float z[32];
    {
        const float* zscr = (const float*)sK;
        #pragma unroll
        for (int b = 0; b < 8; ++b) {
            float4 q = *(const float4*)(zscr + own + b * 4);
            z[b*4+0] = q.x; z[b*4+1] = q.y; z[b*4+2] = q.z; z[b*4+3] = q.w;
        }
    }
    __syncthreads();   // everyone has z; sK can become k-slot storage

    const float H = HSTEP;
    const float C2[1] = {H * 0.2f};
    const float C3[2] = {H * 0.075f, H * 0.225f};
    const float C4[3] = {H * 0.97777777777f, H * -3.73333333333f, H * 3.55555555555f};
    const float C5[4] = {H * 2.95259868922420f, H * -11.5957933241884f,
                         H * 9.82289285169944f, H * -0.29080932784636f};
    const float C6[5] = {H * 2.84627525252525f, H * -10.7575757575758f,
                         H * 8.90642271774347f, H * 0.27840909090909f,
                         H * -0.27351316648377f};
    const float CB[6] = {H * 0.09114583333333f, 0.0f,
                         H * 0.44923629829281f, H * 0.65104166666667f,
                         H * -0.32239527231030f, H * 0.13095238095238f};

    #pragma unroll 1
    for (int step = 0; step < NSTEPS; ++step) {
        // stage 1: arg = z
        store_row_bf16(bufA + own, z);
        feval(bufA, bufB, kb + 0 * 8192, w1p, b1, w2p, b2, w3p, b3, wave, lane);
        // stage 2..6
        make_zarg<1>(bufA + own, kb + own, z, C2);
        feval(bufA, bufB, kb + 1 * 8192, w1p, b1, w2p, b2, w3p, b3, wave, lane);
        make_zarg<2>(bufA + own, kb + own, z, C3);
        feval(bufA, bufB, kb + 2 * 8192, w1p, b1, w2p, b2, w3p, b3, wave, lane);
        make_zarg<3>(bufA + own, kb + own, z, C4);
        feval(bufA, bufB, kb + 3 * 8192, w1p, b1, w2p, b2, w3p, b3, wave, lane);
        make_zarg<4>(bufA + own, kb + own, z, C5);
        feval(bufA, bufB, kb + 4 * 8192, w1p, b1, w2p, b2, w3p, b3, wave, lane);
        make_zarg<5>(bufA + own, kb + own, z, C6);
        feval(bufA, bufB, kb + 5 * 8192, w1p, b1, w2p, b2, w3p, b3, wave, lane);
        // z += h * (b1 k1 + b3 k3 + b4 k4 + b5 k5 + b6 k6)
        update_z<6>(z, kb + own, CB);
        __syncthreads();   // done reading k slots before next step overwrites
    }

    // ---- decoder: (32x256) @ (256x512) + dec_b -> out (global f32) ----
    store_row_bf16(bufA + own, z);
    __syncthreads();
    {
        v8f a0[4], a1[4];
        #pragma unroll
        for (int g = 0; g < 4; ++g) { a0[g] = zero8(); a1[g] = zero8(); }
        int nt0 = wave * 4;
        gemm_acc2<4, 8, LAT>(bufA, decp, nt0, lane, a0, a1);
        const int col16 = lane & 15;
        const int mb    = (lane >> 4) << 3;
        #pragma unroll
        for (int g = 0; g < 4; ++g) {
            const int col = (nt0 + g) * 16 + col16;
            const float bb = dec_b[col];
            #pragma unroll
            for (int r = 0; r < 8; ++r) {
                out[(size_t)(row0 + mb + r) * IN_DIM + col]      = a0[g][r] + bb;
                out[(size_t)(row0 + 16 + mb + r) * IN_DIM + col] = a1[g][r] + bb;
            }
        }
    }
}

// ---------------------------------------------------------------------------
extern "C" void kernel_launch(void* const* d_in, const int* in_sizes, int n_in,
                              void* d_out, int out_size, void* d_ws, size_t ws_size,
                              hipStream_t stream) {
    (void)in_sizes; (void)n_in; (void)out_size; (void)ws_size;
    const float* x     = (const float*)d_in[0];
    const float* enc_w = (const float*)d_in[1];
    const float* enc_b = (const float*)d_in[2];
    const float* w1    = (const float*)d_in[3];
    const float* b1    = (const float*)d_in[4];
    const float* w2    = (const float*)d_in[5];
    const float* b2    = (const float*)d_in[6];
    const float* w3    = (const float*)d_in[7];
    const float* b3    = (const float*)d_in[8];
    const float* dec_w = (const float*)d_in[9];
    const float* dec_b = (const float*)d_in[10];

    unsigned char* ws = (unsigned char*)d_ws;
    __bf16* encp = (__bf16*)(ws + 0);         // 512x256  -> 256 KB
    __bf16* w1p  = (__bf16*)(ws + 262144);    // 256x1024 -> 512 KB
    __bf16* w2p  = (__bf16*)(ws + 786432);    // 1024x1024-> 2 MB
    __bf16* w3p  = (__bf16*)(ws + 2883584);   // 1024x256 -> 512 KB
    __bf16* decp = (__bf16*)(ws + 3407872);   // 256x512  -> 256 KB

    pack_b_kernel<<<(512 * 256) / 256, 256, 0, stream>>>(enc_w, encp, 512, 256);
    pack_b_kernel<<<(256 * 1024) / 256, 256, 0, stream>>>(w1, w1p, 256, 1024);
    pack_b_kernel<<<(1024 * 1024) / 256, 256, 0, stream>>>(w2, w2p, 1024, 1024);
    pack_b_kernel<<<(1024 * 256) / 256, 256, 0, stream>>>(w3, w3p, 1024, 256);
    pack_b_kernel<<<(256 * 512) / 256, 256, 0, stream>>>(dec_w, decp, 256, 512);

    node_fused_kernel<<<B_ROWS / MT, 256, 0, stream>>>(
        x, encp, enc_b, w1p, b1, w2p, b2, w3p, b3, decp, dec_b, (float*)d_out);
}
